// ParametricAttention_71640054497341
// MI455X (gfx1250) — 8-bit matrix operand order repaired and checked by reading, compiled, not run on hardware
//
#include <hip/hip_runtime.h>

typedef __attribute__((ext_vector_type(8))) int v8i;

#define DIN   2048
#define DOUT  2048
#define NP    16
#define NTOK  16384

__global__ __launch_bounds__(256) void k_wscales(const float* __restrict__ Wk,
                                                 const float* __restrict__ Wv,
                                                 float* __restrict__ scales) {
  const float* W = (blockIdx.x == 0) ? Wk : Wv;
  __shared__ float red[256];
  float s = 0.f;
  for (int i = threadIdx.x; i < NP * DIN; i += 256) s += fabsf(W[i]);
  red[threadIdx.x] = s;
  __syncthreads();
  for (int off = 128; off > 0; off >>= 1) {
    if (threadIdx.x < off) red[threadIdx.x] += red[threadIdx.x + off];
    __syncthreads();
  }
  if (threadIdx.x == 0) {
    float m = red[0] * (1.f / (float)(NP * DIN));
    scales[blockIdx.x] = fmaxf(m, 1e-5f);
  }
}

__global__ __launch_bounds__(256) void k_wquant(const float* __restrict__ Wk,
                                                const float* __restrict__ Wv,
                                                const float* __restrict__ scales,
                                                signed char* __restrict__ wkq,
                                                signed char* __restrict__ wvq) {
  int i = blockIdx.x * 256 + threadIdx.x;
  if (i < NP * DIN) {
    float q = rintf(Wk[i] / scales[0]);
    wkq[i] = (signed char)(int)fminf(fmaxf(q, -1.f), 1.f);
  } else {
    int j = i - NP * DIN;
    float q = rintf(Wv[j] / scales[1]);
    wvq[j] = (signed char)(int)fminf(fmaxf(q, -1.f), 1.f);
  }
}

__device__ __forceinline__ int pack4(float4 f, float s) {
  float a = fminf(fmaxf(rintf(f.x * s), -128.f), 127.f);
  float b = fminf(fmaxf(rintf(f.y * s), -128.f), 127.f);
  float c = fminf(fmaxf(rintf(f.z * s), -128.f), 127.f);
  float d = fminf(fmaxf(rintf(f.w * s), -128.f), 127.f);
  return (((int)a) & 255) | ((((int)b) & 255) << 8) |
         ((((int)c) & 255) << 16) | ((((int)d) & 255) << 24);
}

__global__ __launch_bounds__(128) void k_attn(const float* __restrict__ x,
                                              const signed char* __restrict__ wkq,
                                              const float* __restrict__ scales,
                                              signed char* __restrict__ aq,
                                              float* __restrict__ rec2_out) {
  const int lane = threadIdx.x & 31;
  const int wave = threadIdx.x >> 5;
  const int tok0 = blockIdx.x * 16;
  const int half = lane >> 4;
  const int nl   = lane & 15;

  __shared__ float amax_sh[16];
  __shared__ int   acc_sh[4][32][8];

  for (int rr = 0; rr < 4; ++rr) {
    const int m = wave * 4 + rr;
    const float4* row = (const float4*)(x + (size_t)(tok0 + m) * DIN);
    float a = 0.f;
#pragma unroll
    for (int i = 0; i < 16; ++i) {
      float4 v = row[lane + 32 * i];
      a = fmaxf(a, fmaxf(fmaxf(fabsf(v.x), fabsf(v.y)),
                         fmaxf(fabsf(v.z), fabsf(v.w))));
    }
#pragma unroll
    for (int msk = 16; msk >= 1; msk >>= 1) a = fmaxf(a, __shfl_xor(a, msk, 32));
    if (lane == 0) amax_sh[m] = a;
  }
  __syncthreads();

  const float amax_c     = fmaxf(amax_sh[nl], 1e-5f);
  const float xscale_own = 127.f * __builtin_amdgcn_rcpf(amax_c);

  v8i acc = {0, 0, 0, 0, 0, 0, 0, 0};
  const float* xrow = x + (size_t)(tok0 + nl) * DIN;
  const signed char* bk = wkq + nl * DIN;
  const int kend = wave * 512 + 512;
  for (int kb = wave * 512; kb < kend; kb += 64) {
    v8i A;
#pragma unroll
    for (int pr = 0; pr < 4; ++pr) {
      int k0 = kb + (pr >> 1) * 32 + half * 16 + (pr & 1) * 8;
      float4 f0 = *(const float4*)(xrow + k0);
      float4 f1 = *(const float4*)(xrow + k0 + 4);
      A[pr * 2 + 0] = pack4(f0, xscale_own);
      A[pr * 2 + 1] = pack4(f1, xscale_own);
    }
    v8i B;
    int4 b0 = *(const int4*)(bk + kb + half * 16);
    int4 b1 = *(const int4*)(bk + kb + 32 + half * 16);
    B[0] = b0.x; B[1] = b0.y; B[2] = b0.z; B[3] = b0.w;
    B[4] = b1.x; B[5] = b1.y; B[6] = b1.z; B[7] = b1.w;
    acc = __builtin_amdgcn_wmma_i32_16x16x64_iu8(true, A, true, B, acc,
                                                 false, false);
  }
#pragma unroll
  for (int r = 0; r < 8; ++r) acc_sh[wave][lane][r] = acc[r];
  __syncthreads();

  if (wave == 0) {
    const float wk_s      = scales[0];
    const float recip_own = amax_c * (1.f / 127.f);
    float vals[8];
#pragma unroll
    for (int r = 0; r < 8; ++r) {
      int tot = acc_sh[0][lane][r] + acc_sh[1][lane][r] +
                acc_sh[2][lane][r] + acc_sh[3][lane][r];
      int t = r + half * 8;
      float rq = __shfl(recip_own, t, 32);
      float v = (float)tot * wk_s * rq;
      v = 0.5f * v * (1.f + erff(v * 0.70710678118654752f));
      vals[r] = v;
    }
#pragma unroll
    for (int r = 0; r < 8; ++r) {
      int t = r + half * 8;
      float ss = vals[r] * vals[r];
#pragma unroll
      for (int msk = 8; msk >= 1; msk >>= 1) ss += __shfl_xor(ss, msk, 32);
      float norm = fmaxf(sqrtf(ss), 1e-12f);
      float v = vals[r] * __builtin_amdgcn_rcpf(norm) * 4.0f;
      float a = fabsf(v);
#pragma unroll
      for (int msk = 8; msk >= 1; msk >>= 1) a = fmaxf(a, __shfl_xor(a, msk, 32));
      a = fmaxf(a, 1e-5f);
      float q = fminf(fmaxf(rintf(v * 127.f * __builtin_amdgcn_rcpf(a)),
                            -128.f), 127.f);
      aq[(size_t)(tok0 + t) * NP + nl] = (signed char)(int)q;
      if (nl == 0) rec2_out[tok0 + t] = a * (1.f / 127.f);
    }
  }
}

__global__ __launch_bounds__(256) void k_out(const signed char* __restrict__ aq,
                                             const signed char* __restrict__ wvq,
                                             const float* __restrict__ scales,
                                             const float* __restrict__ rec2,
                                             float* __restrict__ out) {
  const int lane = threadIdx.x & 31;
  const int wave = threadIdx.x >> 5;
  const int tok0 = blockIdx.x * 16;
  const int half = lane >> 4;
  const int nl   = lane & 15;

  const float wv_s = scales[1];
  const signed char* arow = aq + (size_t)tok0 * NP;

  v8i A2 = {0, 0, 0, 0, 0, 0, 0, 0};
  if (half == 0) {
    int4 a = *(const int4*)(arow + nl * NP);
    A2[0] = a.x; A2[1] = a.y; A2[2] = a.z; A2[3] = a.w;
  }

  float r2[8];
#pragma unroll
  for (int r = 0; r < 8; ++r)
    r2[r] = rec2[tok0 + r + half * 8] * wv_s;

  const int ct0 = wave * 16;
  for (int ct = ct0; ct < ct0 + 16; ++ct) {
    v8i B2 = {0, 0, 0, 0, 0, 0, 0, 0};
    if (half == 0) {
      int4 b = *(const int4*)(wvq + (size_t)(ct * 16 + nl) * NP);
      B2[0] = b.x; B2[1] = b.y; B2[2] = b.z; B2[3] = b.w;
    }
    v8i c2 = {0, 0, 0, 0, 0, 0, 0, 0};
    c2 = __builtin_amdgcn_wmma_i32_16x16x64_iu8(true, A2, true, B2, c2,
                                                false, false);
#pragma unroll
    for (int r = 0; r < 8; ++r) {
      out[(size_t)(tok0 + r + half * 8) * DOUT + ct * 16 + nl] =
          (float)c2[r] * r2[r];
    }
  }
}

extern "C" void kernel_launch(void* const* d_in, const int* in_sizes, int n_in,
                              void* d_out, int out_size, void* d_ws, size_t ws_size,
                              hipStream_t stream) {
  const float* x  = (const float*)d_in[0];
  const float* Wk = (const float*)d_in[1];
  const float* Wv = (const float*)d_in[2];
  float* out = (float*)d_out;

  float* scales = (float*)d_ws;
  signed char* wkq = (signed char*)d_ws + 256;
  signed char* wvq = wkq + NP * DIN;
  float* rec2      = (float*)((char*)d_ws + 65792);
  signed char* aq  = (signed char*)d_ws + 131328;

  k_wscales<<<2, 256, 0, stream>>>(Wk, Wv, scales);
  k_wquant<<<256, 256, 0, stream>>>(Wk, Wv, scales, wkq, wvq);
  k_attn<<<NTOK / 16, 128, 0, stream>>>(x, wkq, scales, aq, rec2);
  k_out<<<NTOK / 16, 256, 0, stream>>>(aq, wvq, scales, rec2, out);
}
